// CrossDeformableAttentionBlockwFlow_12738873000176
// MI455X (gfx1250) — matrix operand order repaired and checked by reading, not compiled, not run on hardware
//
#include <hip/hip_runtime.h>
#include <math.h>

// ---------------------------------------------------------------------------
// CrossDeformableAttentionBlockwFlow for MI455X (gfx1250, wave32, WMMA)
// B=2, C=96, H=W=96, N_SAMPLES=9, N_GROUPS=12, N_HEADS=12, MLP_HID=192
//
// GEMM-shaped work (3x3 convs as per-tap implicit GEMM, 1x1 projections, MLP)
// runs on v_wmma_f32_16x16x32_bf16:
//   - weights pre-packed to A-fragment order (bf16) -> contiguous b128 loads
//   - input tile staged k-contiguous (bf16) in LDS  -> contiguous ds loads
//   - one block = 64(co) x 32(px) tile; 4 waves share LDS staging; each wave
//     holds 2 accumulators so every A fragment feeds 2 WMMAs.
// ---------------------------------------------------------------------------

typedef __attribute__((ext_vector_type(16))) __bf16 v16bf;
typedef __attribute__((ext_vector_type(8)))  float  v8f;

#define NB      2
#define CC      96
#define HH      96
#define WW      96
#define HWP     (HH * WW)          // 9216
#define NGROUP  12
#define NSAMP   9
#define NHEAD   12
#define DH      8
#define NKV     (2 * NSAMP)        // 18
#define PX      32                 // pixels per block tile (96 % 32 == 0)

// --------------------- weight packing into A-fragment order -----------------
// wpack[((tap*NKC + kci)*NTM + tm)*512 + lane*16 + j]  (bf16, zero padded)
// element (m,k) in this file's contiguous k order (B reads match): m = tm*16 + (lane&15),
// k = kci*32 + j + ((lane>>4)<<4)
__global__ void pack_weights_kernel(const float* __restrict__ w, __bf16* __restrict__ wpack,
                                    int Cin, int Cout, int taps, int nkc, int ntm, long total)
{
    long tid = (long)blockIdx.x * blockDim.x + threadIdx.x;
    if (tid >= total) return;
    int  j    = (int)(tid & 15);
    int  lane = (int)((tid >> 4) & 31);
    long t    = tid >> 9;
    int  tm   = (int)(t % ntm); t /= ntm;
    int  kci  = (int)(t % nkc);
    int  tap  = (int)(t / nkc);
    int  m    = tm * 16 + (lane & 15);
    int  k    = kci * 32 + j + ((lane >> 4) << 4);
    float v   = (m < Cout && k < Cin) ? w[((long)m * Cin + k) * taps + tap] : 0.0f;
    wpack[tid] = (__bf16)v;
}

// ------------------------- WMMA implicit-GEMM conv --------------------------
// grid: (nbm, NTN) ; block: 128.  act: 0 none, 1 PReLU, 2 2*tanh, 3 exact GELU.
template<int CIN, int KS>
__global__ void __launch_bounds__(128)
conv_wmma(const float* __restrict__ in, const __bf16* __restrict__ wpack,
          const float* __restrict__ bias, const float* __restrict__ prelu_a,
          const float* __restrict__ residual, float* __restrict__ out,
          int Cout, int act, long in_bstride, long out_bstride, int ntm)
{
    constexpr int NKC   = (CIN + 31) / 32;
    constexpr int KPAD  = NKC * 32;
    constexpr int HALO  = (KS == 3) ? (PX + 2) : PX;   // 34 or 32 columns
    constexpr int ROWS  = (KS == 3) ? 3 : 1;
    constexpr int SLOTS = ROWS * HALO;                 // 102 or 32
    constexpr int TAPS  = KS * KS;
    __shared__ __attribute__((aligned(64))) __bf16 smem[SLOTS * KPAD];

    int tid    = threadIdx.x;
    int wave   = tid >> 5;
    int lane   = tid & 31;
    int lanehi = lane >> 4;
    int lanelo = lane & 15;

    int bm = blockIdx.x;                // 64-channel block index
    int tn = blockIdx.y;                // 32-pixel tile index
    long p0 = (long)tn * PX;
    int b  = (int)(p0 / HWP);
    int pp = (int)(p0 % HWP);
    int y  = pp / WW;
    int x0 = pp % WW;

    // ---- cooperative stage: input tile, k-contiguous bf16 in LDS ----
    const float* inb = in + (long)b * in_bstride;
    for (int e = tid; e < SLOTS * KPAD; e += 128) {
        int s = e % SLOTS;
        int k = e / SLOTS;
        float v = 0.0f;
        if (k < CIN) {
            if (KS == 3) {
                int ry = s / HALO, cx = s % HALO;
                int gy = y - 1 + ry, gx = x0 - 1 + cx;
                if ((unsigned)gy < (unsigned)HH && (unsigned)gx < (unsigned)WW)
                    v = inb[(long)k * HWP + gy * WW + gx];
            } else {
                v = inb[(long)k * HWP + pp + s];
            }
        }
        smem[s * KPAD + k] = (__bf16)v;
    }
    __syncthreads();

    int tm  = bm * 4 + wave;
    int co0 = tm * 16;

    v8f acc[2];
#pragma unroll
    for (int t = 0; t < 2; ++t)
#pragma unroll
        for (int r = 0; r < 8; ++r) {
            int co = co0 + r + 8 * lanehi;
            acc[t][r] = (co < Cout) ? bias[co] : 0.0f;
        }

    long wstride = (long)ntm * 512;                        // elems per (tap,kci) group
    const __bf16* wp = wpack + (long)tm * 512 + lane * 16;  // this lane's A slot

    for (int tap = 0; tap < TAPS; ++tap) {
        int dy = tap / KS - (KS >> 1);
        int dx = tap % KS - (KS >> 1);
        // B rows for the two 16-pixel groups of this tile
        int s0, s1;
        if (KS == 3) {
            s0 = (dy + 1) * HALO + (lanelo + dx + 1);
            s1 = s0 + 16;
        } else {
            s0 = lanelo;
            s1 = lanelo + 16;
        }
        const __bf16* brow0 = &smem[s0 * KPAD + (lanehi << 4)];
        const __bf16* brow1 = &smem[s1 * KPAD + (lanehi << 4)];
        if (tap + 1 < TAPS)
            __builtin_prefetch(wp + (long)((tap + 1) * NKC) * wstride, 0, 1);
#pragma unroll
        for (int kci = 0; kci < NKC; ++kci) {
            v16bf afrag = *(const v16bf*)(wp + (long)(tap * NKC + kci) * wstride);
            v16bf b0 = *(const v16bf*)(brow0 + kci * 32);
            v16bf b1 = *(const v16bf*)(brow1 + kci * 32);
            acc[0] = __builtin_amdgcn_wmma_f32_16x16x32_bf16(
                         false, afrag, false, b0, (short)0, acc[0], false, false);
            acc[1] = __builtin_amdgcn_wmma_f32_16x16x32_bf16(
                         false, afrag, false, b1, (short)0, acc[1], false, false);
        }
    }

    // ---- epilogue: residual + activation, fp32 store ----
#pragma unroll
    for (int t = 0; t < 2; ++t)
#pragma unroll
        for (int r = 0; r < 8; ++r) {
            int co = co0 + r + 8 * lanehi;
            if (co >= Cout) continue;
            long idx = (long)b * out_bstride + (long)co * HWP + (pp + t * 16 + lanelo);
            float v = acc[t][r];
            if (residual) v += residual[idx];
            if (act == 1)      v = (v >= 0.0f) ? v : prelu_a[co] * v;
            else if (act == 2) v = 2.0f * tanhf(v);
            else if (act == 3) v = 0.5f * v * (1.0f + erff(v * 0.70710678118654752f));
            out[idx] = v;
        }
}

// ---------------------- backward warp + concat packing ----------------------
__global__ void bwarp_pack_kernel(const float* __restrict__ feat_t,
                                  const float* __restrict__ featx,
                                  const float* __restrict__ ftx,
                                  float* __restrict__ cat)
{
    long tid = (long)blockIdx.x * blockDim.x + threadIdx.x;
    long total = (long)NB * HWP;
    if (tid >= total) return;
    int b = (int)(tid / HWP);
    int p = (int)(tid % HWP);
    int y = p / WW, x = p % WW;

    const float* ft = ftx + (long)b * 2 * HWP;
    float fx = ft[p];
    float fy = ft[HWP + p];
    float sx = (float)x + fx;
    float sy = (float)y + fy;
    float fx0 = floorf(sx), fy0 = floorf(sy);
    float wx = sx - fx0, wy = sy - fy0;
    int ix0 = (int)fx0, iy0 = (int)fy0;

    float* catb = cat + (long)b * 194 * HWP;
    const float* img = featx + (long)b * CC * HWP;
    const float* tgt = feat_t + (long)b * CC * HWP;

    auto corner = [&](int ix, int iy, long& off, float& msk) {
        bool val = (ix >= 0 && ix < WW && iy >= 0 && iy < HH);
        int cx = min(max(ix, 0), WW - 1);
        int cy = min(max(iy, 0), HH - 1);
        off = (long)cy * WW + cx;
        msk = val ? 1.0f : 0.0f;
    };
    long o00, o10, o01, o11; float m00, m10, m01, m11;
    corner(ix0,     iy0,     o00, m00);
    corner(ix0 + 1, iy0,     o10, m10);
    corner(ix0,     iy0 + 1, o01, m01);
    corner(ix0 + 1, iy0 + 1, o11, m11);
    float w00 = (1.0f - wx) * (1.0f - wy) * m00;
    float w10 = wx * (1.0f - wy) * m10;
    float w01 = (1.0f - wx) * wy * m01;
    float w11 = wx * wy * m11;

    for (int c = 0; c < CC; ++c) {
        const float* ic = img + (long)c * HWP;
        float v = ic[o00] * w00 + ic[o10] * w10 + ic[o01] * w01 + ic[o11] * w11;
        catb[(long)(96 + c) * HWP + p] = v;
        catb[(long)c * HWP + p] = tgt[(long)c * HWP + p];
    }
    catb[(long)192 * HWP + p] = fx;
    catb[(long)193 * HWP + p] = fy;
}

// ----------------------------- channel copy ---------------------------------
__global__ void copy_chans_kernel(const float* __restrict__ src, float* __restrict__ dst,
                                  long count_per_b, long bstride)
{
    long tid = (long)blockIdx.x * blockDim.x + threadIdx.x;
    long total = (long)NB * count_per_b;
    if (tid >= total) return;
    int b = (int)(tid / count_per_b);
    long i = tid % count_per_b;
    dst[(long)b * bstride + i] = src[(long)b * bstride + i];
}

// ------------------------ deformable reference sampling ---------------------
// sx = x + (off_x + ft_x)/4   (reference grid-math quirk),  sy = y + (off_y + ft_y)
__global__ void ref_sample_kernel(const float* __restrict__ featx,
                                  const float* __restrict__ off,
                                  const float* __restrict__ ftx,
                                  float* __restrict__ kv, int sample_off)
{
    long tid = (long)blockIdx.x * blockDim.x + threadIdx.x;
    long total = (long)NB * NGROUP * NSAMP * HWP;
    if (tid >= total) return;
    int p = (int)(tid % HWP); long r = tid / HWP;
    int s = (int)(r % NSAMP); r /= NSAMP;
    int g = (int)(r % NGROUP);
    int b = (int)(r / NGROUP);
    int y = p / WW, x = p % WW;

    const float* ft = ftx + (long)b * 2 * HWP;
    long ob = ((long)b * (NGROUP * NSAMP * 2) + (long)(g * NSAMP + s) * 2) * HWP;
    float fx = off[ob + p] + ft[p];
    float fy = off[ob + HWP + p] + ft[HWP + p];
    float sx = (float)x + fx * 0.25f;
    float sy = (float)y + fy;

    float fx0 = floorf(sx), fy0 = floorf(sy);
    float wx = sx - fx0, wy = sy - fy0;
    int ix0 = (int)fx0, iy0 = (int)fy0;
    auto corner = [&](int ix, int iy, long& offp, float& msk) {
        bool val = (ix >= 0 && ix < WW && iy >= 0 && iy < HH);
        int cx = min(max(ix, 0), WW - 1);
        int cy = min(max(iy, 0), HH - 1);
        offp = (long)cy * WW + cx;
        msk = val ? 1.0f : 0.0f;
    };
    long o00, o10, o01, o11; float m00, m10, m01, m11;
    corner(ix0,     iy0,     o00, m00);
    corner(ix0 + 1, iy0,     o10, m10);
    corner(ix0,     iy0 + 1, o01, m01);
    corner(ix0 + 1, iy0 + 1, o11, m11);
    float w00 = (1.0f - wx) * (1.0f - wy) * m00;
    float w10 = wx * (1.0f - wy) * m10;
    float w01 = (1.0f - wx) * wy * m01;
    float w11 = wx * wy * m11;

    const float* img = featx + (long)b * CC * HWP;
    int n = sample_off + s;
#pragma unroll
    for (int cc = 0; cc < 8; ++cc) {
        int c = g * 8 + cc;
        const float* ic = img + (long)c * HWP;
        float v = ic[o00] * w00 + ic[o10] * w10 + ic[o01] * w01 + ic[o11] * w11;
        kv[(((long)b * CC + c) * NKV + n) * HWP + p] = v;
    }
}

// ---------------- fused deformable attention (k/v proj + softmax) -----------
__global__ void attn_fused_kernel(const float* __restrict__ q,
                                  const float* __restrict__ kv,
                                  const float* __restrict__ kw, const float* __restrict__ kb,
                                  const float* __restrict__ vw, const float* __restrict__ vb,
                                  float* __restrict__ outbuf)
{
    long tid = (long)blockIdx.x * blockDim.x + threadIdx.x;
    long total = (long)NB * NHEAD * HWP;
    if (tid >= total) return;
    int p = (int)(tid % HWP); long r = tid / HWP;
    int h = (int)(r % NHEAD);
    int b = (int)(r / NHEAD);

    const float* kvb = kv + (long)b * CC * NKV * HWP;
    float qv[DH];
#pragma unroll
    for (int d = 0; d < DH; ++d)
        qv[d] = q[((long)b * CC + h * DH + d) * HWP + p];

    float sc[NKV];
    for (int n = 0; n < NKV; ++n) {
        float kd[DH];
#pragma unroll
        for (int d = 0; d < DH; ++d) kd[d] = kb[h * DH + d];
        for (int c = 0; c < CC; ++c) {
            float kvv = kvb[((long)c * NKV + n) * HWP + p];
#pragma unroll
            for (int d = 0; d < DH; ++d)
                kd[d] += kw[(long)(h * DH + d) * CC + c] * kvv;
        }
        float s = 0.0f;
#pragma unroll
        for (int d = 0; d < DH; ++d) s += qv[d] * kd[d];
        sc[n] = s * 0.35355339059327373f;     // 1/sqrt(8)
    }
    float mx = sc[0];
    for (int n = 1; n < NKV; ++n) mx = fmaxf(mx, sc[n]);
    float sum = 0.0f;
    for (int n = 0; n < NKV; ++n) { sc[n] = expf(sc[n] - mx); sum += sc[n]; }
    float inv = 1.0f / sum;

    float od[DH];
#pragma unroll
    for (int d = 0; d < DH; ++d) od[d] = 0.0f;
    for (int n = 0; n < NKV; ++n) {
        float an = sc[n] * inv;
        float vd[DH];
#pragma unroll
        for (int d = 0; d < DH; ++d) vd[d] = vb[h * DH + d];
        for (int c = 0; c < CC; ++c) {
            float kvv = kvb[((long)c * NKV + n) * HWP + p];
#pragma unroll
            for (int d = 0; d < DH; ++d)
                vd[d] += vw[(long)(h * DH + d) * CC + c] * kvv;
        }
#pragma unroll
        for (int d = 0; d < DH; ++d) od[d] += an * vd[d];
    }
#pragma unroll
    for (int d = 0; d < DH; ++d)
        outbuf[((long)b * CC + h * DH + d) * HWP + p] = od[d];
}

// ------------- transposed conv (stride 2, k4, pad 1) + 2*resize2x -----------
__global__ void upflow_kernel(const float* __restrict__ mov,
                              const float* __restrict__ ftx,
                              const float* __restrict__ fw, const float* __restrict__ fb,
                              float* __restrict__ outp)
{
    const int HO = 2 * HH, WO = 2 * WW;
    long tid = (long)blockIdx.x * blockDim.x + threadIdx.x;
    long total = (long)NB * 2 * HO * WO;
    if (tid >= total) return;
    int X = (int)(tid % WO); long r = tid / WO;
    int Y = (int)(r % HO); r /= HO;
    int oc = (int)(r % 2);
    int b  = (int)(r / 2);

    float acc = fb[oc];
    const float* mb = mov + (long)b * CC * HWP;
    for (int ky = 0; ky < 4; ++ky) {
        int iy2 = Y - 2 + ky;
        if (iy2 < 0 || (iy2 & 1)) continue;
        int iy = iy2 >> 1;
        if (iy >= HH) continue;
        for (int kx = 0; kx < 4; ++kx) {
            int ix2 = X - 2 + kx;
            if (ix2 < 0 || (ix2 & 1)) continue;
            int ix = ix2 >> 1;
            if (ix >= WW) continue;
            long pix = (long)iy * WW + ix;
            int widx = (3 - ky) * 4 + (3 - kx);
            for (int c = 0; c < CC; ++c)
                acc += fw[((long)(c * 2 + oc)) * 16 + widx] * mb[(long)c * HWP + pix];
        }
    }

    float gx = (float)X * (95.0f / 191.0f);
    float gy = (float)Y * (95.0f / 191.0f);
    float fx0 = floorf(gx), fy0 = floorf(gy);
    float wx = gx - fx0, wy = gy - fy0;
    int ix0 = (int)fx0, iy0 = (int)fy0;
    const float* fc = ftx + ((long)b * 2 + oc) * HWP;
    auto samp = [&](int ix, int iy) -> float {
        bool val = (ix >= 0 && ix < WW && iy >= 0 && iy < HH);
        int cx = min(max(ix, 0), WW - 1);
        int cy = min(max(iy, 0), HH - 1);
        return val ? fc[(long)cy * WW + cx] : 0.0f;
    };
    float s = samp(ix0, iy0) * (1.0f - wx) * (1.0f - wy)
            + samp(ix0 + 1, iy0) * wx * (1.0f - wy)
            + samp(ix0, iy0 + 1) * (1.0f - wx) * wy
            + samp(ix0 + 1, iy0 + 1) * wx * wy;
    acc += 2.0f * s;

    outp[(((long)b * 2 + oc) * HO + Y) * WO + X] = acc;
}

// ---------------------------------------------------------------------------
extern "C" void kernel_launch(void* const* d_in, const int* in_sizes, int n_in,
                              void* d_out, int out_size, void* d_ws, size_t ws_size,
                              hipStream_t stream)
{
    const float* feat_t = (const float*)d_in[0];
    const float* feat0  = (const float*)d_in[1];
    const float* feat1  = (const float*)d_in[2];
    const float* ft0    = (const float*)d_in[3];
    const float* ft1    = (const float*)d_in[4];
    const float* const* P = (const float* const*)(d_in + 5);
    enum { CRF1_W = 0, CRF1_B, CRF1_P, CRF2_W, CRF2_B, CRF2_P,
           RB1_W, RB1_B, RB1_P, RB2_W, RB2_B, RB2_P,
           RB3_W, RB3_B, RB3_P, RB4_W, RB4_B, RB4_P,
           RB5_W, RB5_B, RB_P, OFF_W, OFF_B, FLOW_W, FLOW_B,
           Q_W, Q_B, K_W, K_B, V_W, V_B, O_W, O_B,
           FC1_W, FC1_B, FC2_W, FC2_B };

    // ---- workspace carve-up (64B-aligned regions) ----
    char* wsb = (char*)d_ws;
    auto allocB = [&](long bytes) { char* q = wsb; wsb += (bytes + 63) & ~63L; return q; };
    auto allocF = [&](long n) { return (float*)allocB(n * 4); };
    auto allocH = [&](long n) { return (__bf16*)allocB(n * 2); };

    float* cat  = allocF((long)NB * 194 * HWP);
    float* x192 = allocF((long)NB * 192 * HWP);
    float* x96  = allocF((long)NB * CC * HWP);
    float* t1   = allocF((long)NB * CC * HWP);
    float* t2   = allocF((long)NB * CC * HWP);
    float* mov0 = allocF((long)NB * CC * HWP);
    float* mov1 = allocF((long)NB * CC * HWP);
    float* offb = allocF((long)NB * 216 * HWP);
    float* kv   = allocF((long)NB * CC * NKV * HWP);
    float* qb   = allocF((long)NB * CC * HWP);
    float* ao   = allocF((long)NB * CC * HWP);
    float* abuf = allocF((long)NB * CC * HWP);
    float* hdn  = allocF((long)NB * 192 * HWP);

    // packed bf16 weights (A-fragment order)
    auto pack = [&](const float* w, int Cin, int Cout, int taps) {
        int nkc = (Cin + 31) / 32;
        int ntm = 4 * ((Cout + 63) / 64);
        long total = (long)taps * nkc * ntm * 512;
        __bf16* wp = allocH(total);
        pack_weights_kernel<<<(int)((total + 255) / 256), 256, 0, stream>>>(
            w, wp, Cin, Cout, taps, nkc, ntm, total);
        return wp;
    };
    __bf16* pk_crf1 = pack(P[CRF1_W], 194, 192, 9);
    __bf16* pk_crf2 = pack(P[CRF2_W], 192,  96, 9);
    __bf16* pk_rb1  = pack(P[RB1_W],   96,  96, 9);
    __bf16* pk_rb2  = pack(P[RB2_W],   48,  48, 9);
    __bf16* pk_rb3  = pack(P[RB3_W],   96,  96, 9);
    __bf16* pk_rb4  = pack(P[RB4_W],   48,  48, 9);
    __bf16* pk_rb5  = pack(P[RB5_W],   96,  96, 9);
    __bf16* pk_off  = pack(P[OFF_W],   96, 216, 9);
    __bf16* pk_q    = pack(P[Q_W],     96,  96, 1);
    __bf16* pk_o    = pack(P[O_W],     96,  96, 1);
    __bf16* pk_fc1  = pack(P[FC1_W],   96, 192, 1);
    __bf16* pk_fc2  = pack(P[FC2_W],  192,  96, 1);

    float* out_main = (float*)d_out;                       // (B,96,96,96)
    float* out_up0  = out_main + (long)NB * CC * HWP;      // (B,2,192,192)
    float* out_up1  = out_up0 + (long)NB * 2 * 192 * 192;

    const int NTN = NB * HWP / PX;                         // 576 pixel tiles
    auto conv = [&](int CinV, int ks, const float* in, const __bf16* wp,
                    const float* bias, const float* pr, const float* res, float* out,
                    int Cout, int act, long inb, long outb) {
        int nbm = (Cout + 63) / 64;
        int ntm = nbm * 4;
        dim3 grid(nbm, NTN);
        if (ks == 3) {
            if      (CinV == 194) conv_wmma<194, 3><<<grid, 128, 0, stream>>>(in, wp, bias, pr, res, out, Cout, act, inb, outb, ntm);
            else if (CinV == 192) conv_wmma<192, 3><<<grid, 128, 0, stream>>>(in, wp, bias, pr, res, out, Cout, act, inb, outb, ntm);
            else if (CinV ==  96) conv_wmma< 96, 3><<<grid, 128, 0, stream>>>(in, wp, bias, pr, res, out, Cout, act, inb, outb, ntm);
            else                  conv_wmma< 48, 3><<<grid, 128, 0, stream>>>(in, wp, bias, pr, res, out, Cout, act, inb, outb, ntm);
        } else {
            if      (CinV ==  96) conv_wmma< 96, 1><<<grid, 128, 0, stream>>>(in, wp, bias, pr, res, out, Cout, act, inb, outb, ntm);
            else                  conv_wmma<192, 1><<<grid, 128, 0, stream>>>(in, wp, bias, pr, res, out, Cout, act, inb, outb, ntm);
        }
    };
    auto elts = [](long n) { return (int)((n + 255) / 256); };

    const long CB = (long)CC * HWP;

    auto branch = [&](const float* featx, const float* ftx, float* mov,
                      int sample_off, float* up_out) {
        bwarp_pack_kernel<<<elts((long)NB * HWP), 256, 0, stream>>>(feat_t, featx, ftx, cat);
        conv(194, 3, cat,  pk_crf1, P[CRF1_B], P[CRF1_P], nullptr, x192, 192, 1, 194L * HWP, 192L * HWP);
        conv(192, 3, x192, pk_crf2, P[CRF2_B], P[CRF2_P], nullptr, x96,   96, 1, 192L * HWP, CB);
        conv( 96, 3, x96,  pk_rb1,  P[RB1_B],  P[RB1_P],  nullptr, t1,    96, 1, CB, CB);
        copy_chans_kernel<<<elts((long)NB * 48 * HWP), 256, 0, stream>>>(t1, t2, 48L * HWP, CB);
        conv( 48, 3, t1 + 48L * HWP, pk_rb2, P[RB2_B], P[RB2_P], nullptr, t2 + 48L * HWP, 48, 1, CB, CB);
        conv( 96, 3, t2,   pk_rb3,  P[RB3_B],  P[RB3_P],  nullptr, t1,    96, 1, CB, CB);
        copy_chans_kernel<<<elts((long)NB * 48 * HWP), 256, 0, stream>>>(t1, t2, 48L * HWP, CB);
        conv( 48, 3, t1 + 48L * HWP, pk_rb4, P[RB4_B], P[RB4_P], nullptr, t2 + 48L * HWP, 48, 1, CB, CB);
        conv( 96, 3, t2,   pk_rb5,  P[RB5_B],  P[RB_P],   x96,    mov,    96, 1, CB, CB);
        conv( 96, 3, mov,  pk_off,  P[OFF_B],  nullptr,   nullptr, offb, 216, 2, CB, 216L * HWP);
        ref_sample_kernel<<<elts((long)NB * NGROUP * NSAMP * HWP), 256, 0, stream>>>(
            featx, offb, ftx, kv, sample_off);
        upflow_kernel<<<elts((long)NB * 2 * 192 * 192), 256, 0, stream>>>(
            mov, ftx, P[FLOW_W], P[FLOW_B], up_out);
    };

    branch(feat0, ft0, mov0, 0,     out_up0);
    branch(feat1, ft1, mov1, NSAMP, out_up1);

    conv( 96, 1, feat_t, pk_q, P[Q_B], nullptr, nullptr, qb, 96, 0, CB, CB);
    attn_fused_kernel<<<elts((long)NB * NHEAD * HWP), 256, 0, stream>>>(
        qb, kv, P[K_W], P[K_B], P[V_W], P[V_B], ao);
    conv( 96, 1, ao,   pk_o,   P[O_B],   nullptr, nullptr, abuf, 96, 0, CB, CB);
    conv( 96, 1, abuf, pk_fc1, P[FC1_B], nullptr, nullptr, hdn, 192, 3, CB, 192L * HWP);
    conv(192, 1, hdn,  pk_fc2, P[FC2_B], nullptr, abuf, out_main, 96, 0, 192L * HWP, CB);
}
